// MultiHeadAttention_34454227648709
// MI455X (gfx1250) — matrix operand order repaired and checked by reading, not compiled, not run on hardware
//
#include <hip/hip_runtime.h>

#define BB 4
#define TT 2048
#define FFdim 1024
#define HH 16
#define DKK 64

typedef __attribute__((ext_vector_type(16))) __bf16 v16bf;
typedef __attribute__((ext_vector_type(8)))  __bf16 v8bf;
typedef __attribute__((ext_vector_type(4)))  __bf16 v4bf;
typedef __attribute__((ext_vector_type(8)))  float  v8f;
typedef __attribute__((ext_vector_type(4)))  float  v4f;

static __device__ __forceinline__ v16bf bcat(v8bf lo, v8bf hi) {
  return __builtin_shufflevector(lo, hi, 0,1,2,3,4,5,6,7,8,9,10,11,12,13,14,15);
}
static __device__ __forceinline__ v8bf ldg8(const __bf16* p) {
  return *reinterpret_cast<const v8bf*>(p);
}
static __device__ __forceinline__ v8f wmma_bf16(v16bf a, v16bf b, v8f c) {
  return __builtin_amdgcn_wmma_f32_16x16x32_bf16(false, a, false, b, (short)0, c, false, false);
}

// ---------------------------------------------------------------------------
// f32 -> bf16 elementwise convert (vectorized x4)
// ---------------------------------------------------------------------------
__global__ void cvt_f32_to_bf16(const float* __restrict__ src,
                                __bf16* __restrict__ dst, int n) {
  int i = (blockIdx.x * blockDim.x + threadIdx.x) * 4;
  if (i + 3 < n) {
    v4f v = *reinterpret_cast<const v4f*>(src + i);
    v4bf o;
    o[0] = (__bf16)v[0]; o[1] = (__bf16)v[1];
    o[2] = (__bf16)v[2]; o[3] = (__bf16)v[3];
    *reinterpret_cast<v4bf*>(dst + i) = o;
  }
}

// ---------------------------------------------------------------------------
// Prepack K x N f32 weight into B-fragment-major bf16 blocks.
// Block (kt, nt): lane L stores 16 contiguous bf16:
//   element j = W[kt*32 + (L>>4)*8 + (j<8 ? j : 16 + j - 8)][nt*16 + (L&15)]
// Block order: blk = nt * (K/32) + kt, each block = 512 elems.
// ---------------------------------------------------------------------------
__global__ void prepack_weight(const float* __restrict__ W,
                               __bf16* __restrict__ Wp, int K, int N) {
  int tid  = blockIdx.x * blockDim.x + threadIdx.x;
  int lane = tid & 31;
  int blk  = tid >> 5;
  int KT   = K >> 5;
  int kt   = blk % KT;
  int nt   = blk / KT;
  if (nt >= (N >> 4)) return;
  int krow = kt * 32 + (lane >> 4) * 8;
  int col  = nt * 16 + (lane & 15);
  v8bf lo, hi;
#pragma unroll
  for (int j = 0; j < 8; ++j) lo[j] = (__bf16)W[(size_t)(krow + j) * N + col];
#pragma unroll
  for (int j = 0; j < 8; ++j) hi[j] = (__bf16)W[(size_t)(krow + 16 + j) * N + col];
  v8bf* dst = reinterpret_cast<v8bf*>(Wp + (size_t)blk * 512 + lane * 16);
  dst[0] = lo;
  dst[1] = hi;
}

// ---------------------------------------------------------------------------
// C[M,N] = A[M,K](bf16, row-major) @ Wp(prepacked bf16) + bias
// One wave computes a 32x64 tile: 2 A-fragments shared across 4 B-tiles,
// 8 WMMAs per 32-deep K-step; B fragments loaded one at a time.
// MODE 0: bf16 row-major.  MODE 1: bf16 per-head transposed
// (out[(b*N + col)*T + t], for V).  MODE 2: f32 row-major.
// ---------------------------------------------------------------------------
template <int MODE>
__global__ void __launch_bounds__(256, 1)
gemm_wmma(const __bf16* __restrict__ A,
          const __bf16* __restrict__ Wp,
          const float* __restrict__ bias,
          void* __restrict__ out, int M, int N, int K) {
  int wid  = blockIdx.x * (blockDim.x >> 5) + (threadIdx.x >> 5);
  int lane = threadIdx.x & 31;
  int nb64 = N >> 6;
  int mt   = wid / nb64;
  int nb   = wid % nb64;
  if (mt >= (M >> 5)) return;
  int g  = lane >> 4;
  int ln = lane & 15;
  int KT = K >> 5;

  v8f accA0 = {}, accA1 = {}, accA2 = {}, accA3 = {};
  v8f accB0 = {}, accB1 = {}, accB2 = {}, accB3 = {};
  const __bf16* arowA = A + (size_t)(mt * 32 + ln) * K;
  const __bf16* arowB = arowA + (size_t)16 * K;

  for (int kt = 0; kt < KT; ++kt) {
    v16bf afA = bcat(ldg8(arowA + kt * 32 + g * 8),
                     ldg8(arowA + kt * 32 + 16 + g * 8));
    v16bf afB = bcat(ldg8(arowB + kt * 32 + g * 8),
                     ldg8(arowB + kt * 32 + 16 + g * 8));
    const __bf16* wb = Wp + ((size_t)(nb * 4) * KT + kt) * 512 + lane * 16;
    if (kt + 1 < KT) {  // prefetch next K-step (global_prefetch_b8)
      __builtin_prefetch(arowA + (kt + 1) * 32 + g * 8, 0, 1);
      __builtin_prefetch(wb + 512, 0, 1);
    }
    v16bf bf = bcat(ldg8(wb), ldg8(wb + 8));
    accA0 = wmma_bf16(afA, bf, accA0);
    accB0 = wmma_bf16(afB, bf, accB0);
    wb += (size_t)KT * 512;
    bf = bcat(ldg8(wb), ldg8(wb + 8));
    accA1 = wmma_bf16(afA, bf, accA1);
    accB1 = wmma_bf16(afB, bf, accB1);
    wb += (size_t)KT * 512;
    bf = bcat(ldg8(wb), ldg8(wb + 8));
    accA2 = wmma_bf16(afA, bf, accA2);
    accB2 = wmma_bf16(afB, bf, accB2);
    wb += (size_t)KT * 512;
    bf = bcat(ldg8(wb), ldg8(wb + 8));
    accA3 = wmma_bf16(afA, bf, accA3);
    accB3 = wmma_bf16(afB, bf, accB3);
  }

  int n0 = nb * 64;
  float bi0 = bias ? bias[n0 + ln]      : 0.0f;
  float bi1 = bias ? bias[n0 + 16 + ln] : 0.0f;
  float bi2 = bias ? bias[n0 + 32 + ln] : 0.0f;
  float bi3 = bias ? bias[n0 + 48 + ln] : 0.0f;

#pragma unroll
  for (int half = 0; half < 2; ++half) {
    v8f a0 = half ? accB0 : accA0, a1 = half ? accB1 : accA1;
    v8f a2 = half ? accB2 : accA2, a3 = half ? accB3 : accA3;
#pragma unroll
    for (int r = 0; r < 8; ++r) {
      int row = mt * 32 + half * 16 + r + 8 * g;
      float v0 = a0[r] + bi0, v1 = a1[r] + bi1;
      float v2 = a2[r] + bi2, v3 = a3[r] + bi3;
      if constexpr (MODE == 0) {
        __bf16* O = (__bf16*)out;
        size_t base = (size_t)row * N + n0 + ln;
        O[base] = (__bf16)v0; O[base + 16] = (__bf16)v1;
        O[base + 32] = (__bf16)v2; O[base + 48] = (__bf16)v3;
      } else if constexpr (MODE == 1) {
        __bf16* O = (__bf16*)out;
        int bb = row / TT, t = row % TT;
        O[(size_t)(bb * N + n0 + ln) * TT + t]      = (__bf16)v0;
        O[(size_t)(bb * N + n0 + 16 + ln) * TT + t] = (__bf16)v1;
        O[(size_t)(bb * N + n0 + 32 + ln) * TT + t] = (__bf16)v2;
        O[(size_t)(bb * N + n0 + 48 + ln) * TT + t] = (__bf16)v3;
      } else {
        float* O = (float*)out;
        size_t base = (size_t)row * N + n0 + ln;
        O[base] = v0; O[base + 16] = v1; O[base + 32] = v2; O[base + 48] = v3;
      }
    }
  }
}

// ---------------------------------------------------------------------------
// Flash attention: one wave handles 32 query rows of one (b,h) (two 16-row
// C tiles A and B), sharing every K/V fragment between both tiles.
// K fragments are loaded one at a time and consumed immediately by both
// row-tiles to keep peak VGPR liveness low (no scratch spills).
// Row sums come from an extra WMMA against an all-ones B matrix.
// ---------------------------------------------------------------------------
__global__ void __launch_bounds__(256, 1)
flash_attn(const __bf16* __restrict__ Qp,
           const __bf16* __restrict__ Kp,
           const __bf16* __restrict__ Vt,
           const unsigned char* __restrict__ mask,
           __bf16* __restrict__ Xp) {
  // 32 P-rows per wave, row stride 72 bf16 = 144B (16B-aligned, conflict-free)
  __shared__ __bf16 lds[8][32 * 72];
  int wib  = threadIdx.x >> 5;
  int wid  = blockIdx.x * (blockDim.x >> 5) + wib;
  int lane = threadIdx.x & 31;
  int g  = lane >> 4;
  int ln = lane & 15;
  const int MT = TT / 32;
  int bh = wid / MT;
  int mt = wid % MT;
  if (bh >= BB * HH) return;
  int b = bh / HH;
  int h = bh % HH;
  int qbase = mt * 32;

  // Hoisted Q fragments: two 16-row tiles x two 32-deep dk-steps.
  const __bf16* qrA = Qp + ((size_t)(b * TT + qbase + ln) * FFdim + h * DKK);
  const __bf16* qrB = qrA + (size_t)16 * FFdim;
  v16bf qfA0 = bcat(ldg8(qrA + g * 16),      ldg8(qrA + 8 + g * 16));
  v16bf qfA1 = bcat(ldg8(qrA + 32 + g * 16), ldg8(qrA + 40 + g * 16));
  v16bf qfB0 = bcat(ldg8(qrB + g * 16),      ldg8(qrB + 8 + g * 16));
  v16bf qfB1 = bcat(ldg8(qrB + 32 + g * 16), ldg8(qrB + 40 + g * 16));

  v16bf ones;
#pragma unroll
  for (int j = 0; j < 16; ++j) ones[j] = (__bf16)1.0f;

  float mrunA[8], mrunB[8];
#pragma unroll
  for (int r = 0; r < 8; ++r) { mrunA[r] = -3.0e38f; mrunB[r] = -3.0e38f; }
  v8f oA0 = {}, oA1 = {}, oA2 = {}, oA3 = {}, lA = {};
  v8f oB0 = {}, oB1 = {}, oB2 = {}, oB3 = {}, lB = {};

  __bf16* P = &lds[wib][0];
  const float scale = 0.125f;  // 1/sqrt(64)

  for (int k0 = 0; k0 < TT; k0 += 32) {
    // ---- S = Q @ K^T : 32 queries x 32 keys (four 16x16 C tiles).
    //      One K fragment live at a time; each feeds both row tiles. ----
    v8f sA0 = {}, sA1 = {}, sB0 = {}, sB1 = {};
    {
      const __bf16* kr0 = Kp + ((size_t)(b * TT + k0 + ln) * FFdim + h * DKK) + g * 16;
      const __bf16* kr1 = kr0 + (size_t)16 * FFdim;
      v16bf kf = bcat(ldg8(kr0), ldg8(kr0 + 8));          // keys 0-15, dk 0-31
      sA0 = wmma_bf16(qfA0, kf, sA0);
      sB0 = wmma_bf16(qfB0, kf, sB0);
      kf = bcat(ldg8(kr0 + 32), ldg8(kr0 + 40));          // keys 0-15, dk 32-63
      sA0 = wmma_bf16(qfA1, kf, sA0);
      sB0 = wmma_bf16(qfB1, kf, sB0);
      kf = bcat(ldg8(kr1), ldg8(kr1 + 8));                // keys 16-31, dk 0-31
      sA1 = wmma_bf16(qfA0, kf, sA1);
      sB1 = wmma_bf16(qfB0, kf, sB1);
      kf = bcat(ldg8(kr1 + 32), ldg8(kr1 + 40));          // keys 16-31, dk 32-63
      sA1 = wmma_bf16(qfA1, kf, sA1);
      sB1 = wmma_bf16(qfB1, kf, sB1);
    }
    bool m0v = mask[b * TT + k0 + ln] != 0;
    bool m1v = mask[b * TT + k0 + 16 + ln] != 0;

    // ---- online softmax, tile A (LDS rows 0..15) ----
#pragma unroll
    for (int r = 0; r < 8; ++r) {
      float a = m0v ? sA0[r] * scale : -1.0e30f;
      float c = m1v ? sA1[r] * scale : -1.0e30f;
      float mx = fmaxf(a, c);
      mx = fmaxf(mx, __shfl_xor(mx, 1, 32));
      mx = fmaxf(mx, __shfl_xor(mx, 2, 32));
      mx = fmaxf(mx, __shfl_xor(mx, 4, 32));
      mx = fmaxf(mx, __shfl_xor(mx, 8, 32));
      float mnew = fmaxf(mrunA[r], mx);
      float corr = __expf(mrunA[r] - mnew);
      mrunA[r] = mnew;
      float p0 = m0v ? __expf(a - mnew) : 0.0f;
      float p1 = m1v ? __expf(c - mnew) : 0.0f;
      oA0[r] *= corr; oA1[r] *= corr; oA2[r] *= corr; oA3[r] *= corr;
      lA[r] *= corr;
      int row = r + 8 * g;
      P[row * 72 + ln]      = (__bf16)p0;
      P[row * 72 + 16 + ln] = (__bf16)p1;
    }
    // ---- online softmax, tile B (LDS rows 16..31) ----
#pragma unroll
    for (int r = 0; r < 8; ++r) {
      float a = m0v ? sB0[r] * scale : -1.0e30f;
      float c = m1v ? sB1[r] * scale : -1.0e30f;
      float mx = fmaxf(a, c);
      mx = fmaxf(mx, __shfl_xor(mx, 1, 32));
      mx = fmaxf(mx, __shfl_xor(mx, 2, 32));
      mx = fmaxf(mx, __shfl_xor(mx, 4, 32));
      mx = fmaxf(mx, __shfl_xor(mx, 8, 32));
      float mnew = fmaxf(mrunB[r], mx);
      float corr = __expf(mrunB[r] - mnew);
      mrunB[r] = mnew;
      float p0 = m0v ? __expf(a - mnew) : 0.0f;
      float p1 = m1v ? __expf(c - mnew) : 0.0f;
      oB0[r] *= corr; oB1[r] *= corr; oB2[r] *= corr; oB3[r] *= corr;
      lB[r] *= corr;
      int row = 16 + r + 8 * g;
      P[row * 72 + ln]      = (__bf16)p0;
      P[row * 72 + 16 + ln] = (__bf16)p1;
    }

    // ---- P A-fragments from LDS ----
    v16bf pfA = bcat(*(const v8bf*)&P[ln * 72 + g * 16],
                     *(const v8bf*)&P[ln * 72 + 8 + g * 16]);
    v16bf pfB = bcat(*(const v8bf*)&P[(16 + ln) * 72 + g * 16],
                     *(const v8bf*)&P[(16 + ln) * 72 + 8 + g * 16]);

    // ---- running row sums on the matrix unit: l += P @ ones ----
    lA = wmma_bf16(pfA, ones, lA);
    lB = wmma_bf16(pfB, ones, lB);

    // ---- O += P @ V (V fragments shared between both row tiles) ----
    {
      const __bf16* vb = Vt + ((size_t)((b * HH + h) * DKK + ln) * TT + k0 + g * 16);
      v16bf vf = bcat(ldg8(vb), ldg8(vb + 8));
      oA0 = wmma_bf16(pfA, vf, oA0);
      oB0 = wmma_bf16(pfB, vf, oB0);
      vb += (size_t)16 * TT;
      vf = bcat(ldg8(vb), ldg8(vb + 8));
      oA1 = wmma_bf16(pfA, vf, oA1);
      oB1 = wmma_bf16(pfB, vf, oB1);
      vb += (size_t)16 * TT;
      vf = bcat(ldg8(vb), ldg8(vb + 8));
      oA2 = wmma_bf16(pfA, vf, oA2);
      oB2 = wmma_bf16(pfB, vf, oB2);
      vb += (size_t)16 * TT;
      vf = bcat(ldg8(vb), ldg8(vb + 8));
      oA3 = wmma_bf16(pfA, vf, oA3);
      oB3 = wmma_bf16(pfB, vf, oB3);
    }
  }

  // ---- normalize and write x (bf16, (B,T,F) row-major) ----
#pragma unroll
  for (int r = 0; r < 8; ++r) {
    int trow = qbase + r + 8 * g;
    float inv = 1.0f / lA[r];
    size_t base = (size_t)(b * TT + trow) * FFdim + h * DKK + ln;
    Xp[base]      = (__bf16)(oA0[r] * inv);
    Xp[base + 16] = (__bf16)(oA1[r] * inv);
    Xp[base + 32] = (__bf16)(oA2[r] * inv);
    Xp[base + 48] = (__bf16)(oA3[r] * inv);
  }
#pragma unroll
  for (int r = 0; r < 8; ++r) {
    int trow = qbase + 16 + r + 8 * g;
    float inv = 1.0f / lB[r];
    size_t base = (size_t)(b * TT + trow) * FFdim + h * DKK + ln;
    Xp[base]      = (__bf16)(oB0[r] * inv);
    Xp[base + 16] = (__bf16)(oB1[r] * inv);
    Xp[base + 32] = (__bf16)(oB2[r] * inv);
    Xp[base + 48] = (__bf16)(oB3[r] * inv);
  }
}

// ---------------------------------------------------------------------------
extern "C" void kernel_launch(void* const* d_in, const int* in_sizes, int n_in,
                              void* d_out, int out_size, void* d_ws, size_t ws_size,
                              hipStream_t stream) {
  (void)in_sizes; (void)n_in; (void)out_size; (void)ws_size;
  const float* q  = (const float*)d_in[0];
  const float* k  = (const float*)d_in[1];
  const float* v  = (const float*)d_in[2];
  const unsigned char* mask = (const unsigned char*)d_in[3];
  const float* Wq = (const float*)d_in[4];
  const float* bq = (const float*)d_in[5];
  const float* Wk = (const float*)d_in[6];
  const float* bk = (const float*)d_in[7];
  const float* Wv = (const float*)d_in[8];
  const float* bv = (const float*)d_in[9];
  const float* Wo = (const float*)d_in[10];

  const size_t NA = (size_t)BB * TT * FFdim;  // 8,388,608 activations
  const size_t NW = (size_t)FFdim * FFdim;    // 1,048,576 weight elems

  __bf16* ws  = (__bf16*)d_ws;
  __bf16* abf = ws;            // transient bf16 activations (reused as Xp)
  __bf16* Qp  = ws + NA;
  __bf16* Kp  = ws + 2 * NA;
  __bf16* Vtb = ws + 3 * NA;   // V transposed (B,H,DK,T)
  __bf16* WqP = ws + 4 * NA;
  __bf16* WkP = WqP + NW;
  __bf16* WvP = WkP + NW;
  __bf16* WoP = WvP + NW;
  __bf16* Xp  = abf;

  dim3 blk(256);
  const int ppBlocks   = ((FFdim / 32) * (FFdim / 16) * 32) / 256;  // 256
  const int cvtBlocks  = (int)(NA / 4 / 256);                       // 8192
  const int gemmBlocks = ((BB * TT / 32) * (FFdim / 64)) / 8;       // 512
  const int attnBlocks = (BB * HH * (TT / 32)) / 8;                 // 512

  prepack_weight<<<ppBlocks, blk, 0, stream>>>(Wq, WqP, FFdim, FFdim);
  prepack_weight<<<ppBlocks, blk, 0, stream>>>(Wk, WkP, FFdim, FFdim);
  prepack_weight<<<ppBlocks, blk, 0, stream>>>(Wv, WvP, FFdim, FFdim);
  prepack_weight<<<ppBlocks, blk, 0, stream>>>(Wo, WoP, FFdim, FFdim);

  cvt_f32_to_bf16<<<cvtBlocks, blk, 0, stream>>>(q, abf, (int)NA);
  gemm_wmma<0><<<gemmBlocks, blk, 0, stream>>>(abf, WqP, bq, Qp, BB * TT, FFdim, FFdim);

  cvt_f32_to_bf16<<<cvtBlocks, blk, 0, stream>>>(k, abf, (int)NA);
  gemm_wmma<0><<<gemmBlocks, blk, 0, stream>>>(abf, WkP, bk, Kp, BB * TT, FFdim, FFdim);

  cvt_f32_to_bf16<<<cvtBlocks, blk, 0, stream>>>(v, abf, (int)NA);
  gemm_wmma<1><<<gemmBlocks, blk, 0, stream>>>(abf, WvP, bv, Vtb, BB * TT, FFdim, FFdim);

  flash_attn<<<attnBlocks, blk, 0, stream>>>(Qp, Kp, Vtb, mask, Xp);

  gemm_wmma<2><<<gemmBlocks, blk, 0, stream>>>(Xp, WoP, nullptr, d_out, BB * TT, FFdim, FFdim);
}
